// RelPositionMultiHeadAttention_37769942401601
// MI455X (gfx1250) — matrix operand order repaired and checked by reading, not compiled, not run on hardware
//
#include <hip/hip_runtime.h>
#include <hip/hip_bf16.h>

// ---------------------------------------------------------------------------
// RelPositionMultiHeadAttention for MI455X (gfx1250, wave32, WMMA bf16)
//   H=8, DK=64, F=512, B=8, T=1024, POS=2047
// All matmuls (projections, q_u*K^T, q_v*P^T band, P*V, output proj) run on
// v_wmma_f32_16x16x32_bf16 with f32 accumulation.
// GEMM waves compute 32x64 register tiles: 8 WMMAs per 12 b128 loads.
// ---------------------------------------------------------------------------

typedef __bf16 bf16_t;
typedef __attribute__((ext_vector_type(16))) __bf16 v16bf;
typedef __attribute__((ext_vector_type(8)))  __bf16 v8bf;
typedef __attribute__((ext_vector_type(8)))  float  v8f;

#define B_   8
#define T_   1024
#define F_   512
#define H_   8
#define DK_  64
#define POS_ 2047
#define SCALE_ 0.125f   // 1/sqrt(64)

__device__ __forceinline__ v8f wmma_bf16(v16bf a, v16bf b, v8f c) {
  // (neg_a, A, neg_b, B, c_mod, C, reuse_a, reuse_b)
  return __builtin_amdgcn_wmma_f32_16x16x32_bf16(false, a, false, b, (short)0, c,
                                                 false, false);
}

// A-operand fragment (16x32, M x K) from a row-major tile.
// lanes 0-15 : row = lane,    K = {k0+0..7,  k0+16..23}
// lanes 16-31: row = lane-16, K = {k0+8..15, k0+24..31}
__device__ __forceinline__ v16bf load_afrag(const bf16_t* tile, int stride,
                                            int lane, int k0) {
  int g = lane >> 4, r = lane & 15;
  const bf16_t* p = tile + (size_t)r * stride + k0 + g * 8;
  v8bf lo = *(const v8bf*)(p);
  v8bf hi = *(const v8bf*)(p + 16);
  return __builtin_shufflevector(lo, hi, 0,1,2,3,4,5,6,7,8,9,10,11,12,13,14,15);
}

// B-operand fragment (32x16, K x N). Caller supplies the per-lane "column
// source row" pointer (row-major data whose row index is this lane's N).
// lanes 0-15 hold K = {k0+0..7, k0+16..23}, lanes 16-31 hold K = {k0+8..15, k0+24..31}.
__device__ __forceinline__ v16bf load_bfrag(const bf16_t* rowptr, int lane,
                                            int k0) {
  int g = lane >> 4;
  const bf16_t* p = rowptr + k0 + g * 8;
  v8bf lo = *(const v8bf*)(p);
  v8bf hi = *(const v8bf*)(p + 16);
  return __builtin_shufflevector(lo, hi, 0,1,2,3,4,5,6,7,8,9,10,11,12,13,14,15);
}

// ---------------------------------------------------------------------------
__global__ void f32_to_bf16_k(const float* __restrict__ in,
                              bf16_t* __restrict__ out, long n) {
  long i = (long)blockIdx.x * blockDim.x + threadIdx.x;
  if (i < n) out[i] = (bf16_t)in[i];
}

// ---------------------------------------------------------------------------
// Generic projection GEMM:  Y[m,n] = sum_k A[m,k] * W[n,k]  (+bias)
// One wave computes a 32x64 register tile (2 M-subtiles x 4 N-subtiles);
// the two A fragments are reused across the 4 N-subtiles and the 4 B
// fragments across the 2 M-subtiles: 8 WMMAs per 12 b128 loads per k-step.
// 4 waves/block span 128 rows.
// mode 0: write Qu=(y+bq+pbu), Qv=(y+bq+pbv) into (B,H,T,DK) bf16
// mode 1: write K  into (B,H,T,DK) bf16
// mode 2: write V  into (B,H,DK,T) bf16 (transposed for PV B-operand)
// mode 3: write P  into (H,POS,DK) bf16 (guard m<POS)
// mode 4: write f32 (m,n) row-major + bias (final output)
__global__ __launch_bounds__(128) void gemm_wmma_k(
    const bf16_t* __restrict__ A, const bf16_t* __restrict__ W,
    const float* __restrict__ bias,
    const float* __restrict__ pbu, const float* __restrict__ pbv,
    bf16_t* __restrict__ out0, bf16_t* __restrict__ out1,
    float* __restrict__ fout, int mode) {
  int lane = threadIdx.x & 31;
  int wave = threadIdx.x >> 5;
  int m0 = (blockIdx.y * 4 + wave) * 32;
  int n0 = blockIdx.x * 64;
  int g = lane >> 4, lr = lane & 15;

  const bf16_t* atile0 = A + (size_t)m0 * F_;
  const bf16_t* atile1 = A + (size_t)(m0 + 16) * F_;
  const bf16_t* wrow0 = W + (size_t)(n0 + lr) * F_;

  v8f acc[2][4] = {};
  for (int k0 = 0; k0 < F_; k0 += 32) {
    v16bf a0 = load_afrag(atile0, F_, lane, k0);
    v16bf a1 = load_afrag(atile1, F_, lane, k0);
#pragma unroll
    for (int n = 0; n < 4; ++n) {
      v16bf bfg = load_bfrag(wrow0 + (size_t)(n * 16) * F_, lane, k0);
      acc[0][n] = wmma_bf16(a0, bfg, acc[0][n]);
      acc[1][n] = wmma_bf16(a1, bfg, acc[1][n]);
    }
  }

#pragma unroll
  for (int mi = 0; mi < 2; ++mi) {
#pragma unroll
    for (int n = 0; n < 4; ++n) {
      int col = n0 + n * 16 + lr;
      float bval = bias ? bias[col] : 0.0f;
      int h = col >> 6, d = col & 63;
#pragma unroll
      for (int r = 0; r < 8; ++r) {
        int m = m0 + mi * 16 + r + g * 8;
        float y = acc[mi][n][r] + bval;
        if (mode == 0) {
          int b = m >> 10, t = m & (T_ - 1);
          size_t idx = (((size_t)(b * H_ + h)) * T_ + t) * DK_ + d;
          out0[idx] = (bf16_t)(y + pbu[col]);
          out1[idx] = (bf16_t)(y + pbv[col]);
        } else if (mode == 1) {
          int b = m >> 10, t = m & (T_ - 1);
          out0[(((size_t)(b * H_ + h)) * T_ + t) * DK_ + d] = (bf16_t)y;
        } else if (mode == 2) {
          int b = m >> 10, t = m & (T_ - 1);
          out0[(((size_t)(b * H_ + h)) * DK_ + d) * T_ + t] = (bf16_t)y;
        } else if (mode == 3) {
          if (m < POS_) out0[((size_t)h * POS_ + m) * DK_ + d] = (bf16_t)y;
        } else {
          fout[(size_t)m * F_ + col] = y;
        }
      }
    }
  }
}

// ---------------------------------------------------------------------------
// Fused relative-position flash attention.
// Grid: (B*H, T/16), block = 32 (one wave). Each block owns 16 query rows.
// Keys processed in chunks of 32 (two 16-wide score subtiles).
__global__ __launch_bounds__(32) void relattn_k(
    const bf16_t* __restrict__ Qu, const bf16_t* __restrict__ Qv,
    const bf16_t* __restrict__ Kb, const bf16_t* __restrict__ Vt,
    const bf16_t* __restrict__ Pb, const unsigned char* __restrict__ mask,
    bf16_t* __restrict__ aout) {
  __shared__ float lds_bd[16][33];   // rel-shift band tile (16 x 32)
  __shared__ float lds_p[16][33];    // probability tile    (16 x 32)

  int lane = threadIdx.x & 31;
  int g = lane >> 4, lr = lane & 15;
  int bh = blockIdx.x;
  int b = bh >> 3, h = bh & 7;
  int q0 = blockIdx.y * 16;

  const bf16_t* Qub = Qu + ((size_t)bh * T_ + q0) * DK_;
  const bf16_t* Qvb = Qv + ((size_t)bh * T_ + q0) * DK_;
  const bf16_t* Kbase = Kb + (size_t)bh * T_ * DK_;
  const bf16_t* Vbase = Vt + (size_t)bh * DK_ * T_;
  const bf16_t* Pbase = Pb + (size_t)h * POS_ * DK_;

  v16bf qu[2], qv[2];
#pragma unroll
  for (int s = 0; s < 2; ++s) {
    qu[s] = load_afrag(Qub, DK_, lane, 32 * s);
    qv[s] = load_afrag(Qvb, DK_, lane, 32 * s);
  }

  float m_i[8], l_i[8];
#pragma unroll
  for (int r = 0; r < 8; ++r) { m_i[r] = -1.0e30f; l_i[r] = 0.0f; }
  v8f o[4] = {};   // 16 x 64 output accumulator (4 N-chunks of 16)

  for (int j0 = 0; j0 < T_; j0 += 32) {
    if (j0 + 32 < T_)
      __builtin_prefetch((const void*)(Kbase + (size_t)(j0 + 32) * DK_), 0, 1);

    v8f sc[2];
#pragma unroll
    for (int s = 0; s < 2; ++s) {
      int jstart = j0 + s * 16;
      // ---- content scores: Qu (16x64) @ K_tile^T (64x16) ----
      const bf16_t* krow = Kbase + (size_t)(jstart + lr) * DK_;
      v8f a = {};
      a = wmma_bf16(qu[0], load_bfrag(krow, lane, 0), a);
      a = wmma_bf16(qu[1], load_bfrag(krow, lane, 32), a);

      // ---- positional band: Qv (16x64) @ Pband^T (64x32) ----
      // bd[qi, ji] needs pos row (jstart+ji) - (q0+qi) + T-1 = base + ji - qi.
      int base = jstart - q0 + (T_ - 1);
      int r0 = base - 15 + lr;  r0 = r0 < 0 ? 0 : (r0 > POS_ - 1 ? POS_ - 1 : r0);
      int r1 = base + 1 + lr;   r1 = r1 < 0 ? 0 : (r1 > POS_ - 1 ? POS_ - 1 : r1);
      const bf16_t* p0 = Pbase + (size_t)r0 * DK_;
      const bf16_t* p1 = Pbase + (size_t)r1 * DK_;
      v8f t0 = {}, t1 = {};
      t0 = wmma_bf16(qv[0], load_bfrag(p0, lane, 0), t0);
      t0 = wmma_bf16(qv[1], load_bfrag(p0, lane, 32), t0);
      t1 = wmma_bf16(qv[0], load_bfrag(p1, lane, 0), t1);
      t1 = wmma_bf16(qv[1], load_bfrag(p1, lane, 32), t1);

      // stage band tile to LDS, gather shifted diagonal
      __syncthreads();
#pragma unroll
      for (int r = 0; r < 8; ++r) {
        lds_bd[r + 8 * g][lr]      = t0[r];
        lds_bd[r + 8 * g][16 + lr] = t1[r];
      }
      __syncthreads();
#pragma unroll
      for (int r = 0; r < 8; ++r) {
        int qi = r + 8 * g;
        float bd = lds_bd[qi][lr - qi + 15];          // c in [0,30]
        float v = (a[r] + bd) * SCALE_;
        const unsigned char mb =
            mask[((size_t)b * T_ + (q0 + qi)) * T_ + (jstart + lr)];
        a[r] = mb ? -1.0e9f : v;
      }
      sc[s] = a;
    }

    // ---- online softmax over the 32-key chunk ----
    float alpha[8];
#pragma unroll
    for (int r = 0; r < 8; ++r) {
      float mx = fmaxf(sc[0][r], sc[1][r]);
#pragma unroll
      for (int off = 8; off >= 1; off >>= 1)
        mx = fmaxf(mx, __shfl_xor(mx, off, 32));
      float mnew = fmaxf(m_i[r], mx);
      float p0 = __expf(sc[0][r] - mnew);
      float p1 = __expf(sc[1][r] - mnew);
      sc[0][r] = p0; sc[1][r] = p1;
      float rs = p0 + p1;
#pragma unroll
      for (int off = 8; off >= 1; off >>= 1)
        rs += __shfl_xor(rs, off, 32);
      alpha[r] = __expf(m_i[r] - mnew);
      l_i[r] = l_i[r] * alpha[r] + rs;
      m_i[r] = mnew;
    }
#pragma unroll
    for (int n = 0; n < 4; ++n)
#pragma unroll
      for (int r = 0; r < 8; ++r) o[n][r] *= alpha[r];

    // ---- restage probabilities C-layout -> A-fragment layout via LDS ----
    __syncthreads();
#pragma unroll
    for (int r = 0; r < 8; ++r) {
      lds_p[r + 8 * g][lr]      = sc[0][r];
      lds_p[r + 8 * g][16 + lr] = sc[1][r];
    }
    __syncthreads();
    v16bf pf;
#pragma unroll
    for (int e = 0; e < 8; ++e) {
      pf[e]     = (bf16_t)lds_p[lr][g * 8 + e];
      pf[e + 8] = (bf16_t)lds_p[lr][g * 8 + 16 + e];
    }

    // ---- O += P (16x32) @ V (32x64), V stored transposed (DK,T) ----
#pragma unroll
    for (int n = 0; n < 4; ++n) {
      const bf16_t* vrow = Vbase + (size_t)(n * 16 + lr) * T_ + j0;
      o[n] = wmma_bf16(pf, load_bfrag(vrow, lane, 0), o[n]);
    }
  }

  // ---- normalize and emit bf16 activations (B,T,F) for output proj ----
#pragma unroll
  for (int n = 0; n < 4; ++n) {
#pragma unroll
    for (int r = 0; r < 8; ++r) {
      int qi = r + 8 * g;
      int d = n * 16 + lr;
      float val = o[n][r] / l_i[r];
      aout[((size_t)b * T_ + (q0 + qi)) * F_ + h * DK_ + d] = (bf16_t)val;
    }
  }
}

// ---------------------------------------------------------------------------
extern "C" void kernel_launch(void* const* d_in, const int* in_sizes, int n_in,
                              void* d_out, int out_size, void* d_ws,
                              size_t ws_size, hipStream_t stream) {
  (void)in_sizes; (void)n_in; (void)out_size; (void)ws_size;
  const float* x    = (const float*)d_in[0];
  const float* pe   = (const float*)d_in[1];
  const unsigned char* mask = (const unsigned char*)d_in[2];
  const float* Wq   = (const float*)d_in[3];
  const float* bq   = (const float*)d_in[4];
  const float* Wk   = (const float*)d_in[5];
  const float* bk   = (const float*)d_in[6];
  const float* Wv   = (const float*)d_in[7];
  const float* bv   = (const float*)d_in[8];
  const float* Wpos = (const float*)d_in[9];
  const float* pbu  = (const float*)d_in[10];
  const float* pbv  = (const float*)d_in[11];
  const float* Wout = (const float*)d_in[12];
  const float* bout = (const float*)d_in[13];
  float* out = (float*)d_out;

  char* ws = (char*)d_ws;
  size_t off = 0;
  auto alloc = [&](size_t bytes) -> char* {
    char* p = ws + off;
    off = (off + bytes + 255) & ~(size_t)255;
    return p;
  };
  const size_t MT = (size_t)B_ * T_;          // 8192
  bf16_t* xb    = (bf16_t*)alloc(MT * F_ * 2);
  bf16_t* peb   = (bf16_t*)alloc((size_t)2048 * F_ * 2);   // padded to 2048 rows
  bf16_t* wqb   = (bf16_t*)alloc((size_t)F_ * F_ * 2);
  bf16_t* wkb   = (bf16_t*)alloc((size_t)F_ * F_ * 2);
  bf16_t* wvb   = (bf16_t*)alloc((size_t)F_ * F_ * 2);
  bf16_t* wposb = (bf16_t*)alloc((size_t)F_ * F_ * 2);
  bf16_t* woutb = (bf16_t*)alloc((size_t)F_ * F_ * 2);
  bf16_t* Qub   = (bf16_t*)alloc(MT * F_ * 2);
  bf16_t* Qvb   = (bf16_t*)alloc(MT * F_ * 2);
  bf16_t* Kbw   = (bf16_t*)alloc(MT * F_ * 2);
  bf16_t* Vtw   = (bf16_t*)alloc(MT * F_ * 2);
  bf16_t* Pbw   = (bf16_t*)alloc((size_t)H_ * POS_ * DK_ * 2);
  bf16_t* aout  = (bf16_t*)alloc(MT * F_ * 2);

  auto cvt = [&](const float* src, bf16_t* dst, long n) {
    f32_to_bf16_k<<<(unsigned)((n + 255) / 256), 256, 0, stream>>>(src, dst, n);
  };
  cvt(x, xb, (long)MT * F_);
  cvt(pe, peb, (long)POS_ * F_);
  cvt(Wq, wqb, (long)F_ * F_);
  cvt(Wk, wkb, (long)F_ * F_);
  cvt(Wv, wvb, (long)F_ * F_);
  cvt(Wpos, wposb, (long)F_ * F_);
  cvt(Wout, woutb, (long)F_ * F_);

  dim3 gblk(128);
  // Q (+rel-pos biases) -> Qu/Qv
  gemm_wmma_k<<<dim3(F_ / 64, MT / 128), gblk, 0, stream>>>(
      xb, wqb, bq, pbu, pbv, Qub, Qvb, nullptr, 0);
  // K
  gemm_wmma_k<<<dim3(F_ / 64, MT / 128), gblk, 0, stream>>>(
      xb, wkb, bk, nullptr, nullptr, Kbw, nullptr, nullptr, 1);
  // V (transposed layout)
  gemm_wmma_k<<<dim3(F_ / 64, MT / 128), gblk, 0, stream>>>(
      xb, wvb, bv, nullptr, nullptr, Vtw, nullptr, nullptr, 2);
  // P (pos projection, no bias), 2048-row padded grid with m<POS guard
  gemm_wmma_k<<<dim3(F_ / 64, 2048 / 128), gblk, 0, stream>>>(
      peb, wposb, nullptr, nullptr, nullptr, Pbw, nullptr, nullptr, 3);
  // fused rel-position flash attention
  relattn_k<<<dim3(B_ * H_, T_ / 16), 32, 0, stream>>>(
      Qub, Qvb, Kbw, Vtw, Pbw, mask, aout);
  // output projection, f32 result
  gemm_wmma_k<<<dim3(F_ / 64, MT / 128), gblk, 0, stream>>>(
      aout, woutb, bout, nullptr, nullptr, nullptr, nullptr, out, 4);
}
